// AutoCorrelation_6390911337306
// MI455X (gfx1250) — matrix operand order repaired and checked by reading, not compiled, not run on hardware
//
#include <hip/hip_runtime.h>
#include <hip/hip_bf16.h>
#include <math.h>
#include <stdint.h>

// Problem constants: B=16, L=3072, H=8, E=64 -> C = H*E = 512, L/16 = 192 tiles.
#define B_   16
#define L_   3072
#define C_   512
#define NT   192
#define TOPK 8

#define SI_TILES 3    // Q tiles staged per block (48 rows)
#define NSI      64   // NT / SI_TILES
#define APAD     520  // padded LDS row stride in elements (512 + 8)

typedef __attribute__((ext_vector_type(16))) __bf16 v16bf;
typedef __attribute__((ext_vector_type(8)))  float  v8f;

// TDM availability / arity discrimination (probe-verified in bridge doc):
// builtin exists on both toolchains; 6-arg form iff therock TDM header ships.
#if defined(__has_builtin)
#  if __has_builtin(__builtin_amdgcn_tensor_load_to_lds)
#    define HAVE_TDM 1
#  endif
#endif
#if defined(__has_include)
#  if __has_include(<hip/amd_detail/amd_gfx1250_TDM.h>)
#    define TDM_6ARG 1
#  endif
#endif

// ---------------------------------------------------------------------------
// RNE float -> bf16 (bit-level)
// ---------------------------------------------------------------------------
__device__ __forceinline__ unsigned short f32_to_bf16(float f) {
  unsigned int u = __float_as_uint(f);
  u += 0x7fffu + ((u >> 16) & 1u);
  return (unsigned short)(u >> 16);
}

// ---------------------------------------------------------------------------
// Kernel 1: convert Q and K (f32, [B][L][C]) to bf16 workspace, same layout.
// ---------------------------------------------------------------------------
__global__ void cvt_bf16_kernel(const float* __restrict__ Q,
                                const float* __restrict__ K,
                                unsigned short* __restrict__ Qbf,
                                unsigned short* __restrict__ Kbf) {
  const float* __restrict__ src = blockIdx.y ? K : Q;
  unsigned short* __restrict__ dst = blockIdx.y ? Kbf : Qbf;
  size_t i = ((size_t)blockIdx.x * blockDim.x + threadIdx.x) * 4;
  float4 f = *(const float4*)(src + i);
  union { unsigned short h[4]; uint2 u; } o;
  o.h[0] = f32_to_bf16(f.x);
  o.h[1] = f32_to_bf16(f.y);
  o.h[2] = f32_to_bf16(f.z);
  o.h[3] = f32_to_bf16(f.w);
  *(uint2*)(dst + i) = o.u;
}

// ---------------------------------------------------------------------------
// Kernel 2: WMMA correlation, LDS-blocked.
//
// Block = 128 threads (4 waves), grid (64, 16) = (si, b).
//  - TDM stages Q rows [48*si, 48*si+48) x 512ch into LDS with padded rows
//    (bank-conflict-free), shared by all 4 waves.
//  - Wave w loops over 24 J-groups, owning K tiles {8*sjg+2w, +1} (regs).
//  - Per k-chunk: 6 WMMAs (3 Q tiles x 2 K tiles) into 4 diagonal-grouped
//    accumulators (tiles with equal i-jj share tau range -> same acc).
//  - Diagonals tau = 16*(Itile-Jtile) + r - n binned into LDS via ds_add_f32,
//    flushed once per block to global mean_value[b][tau].
// ---------------------------------------------------------------------------
__global__ __launch_bounds__(128) void corr_wmma_kernel(
    const unsigned short* __restrict__ Qbf,
    const unsigned short* __restrict__ Kbf,
    float* __restrict__ meanv) {
  __shared__ __align__(16) unsigned short As[SI_TILES * 16 * APAD];  // 49,920 B
  __shared__ float mbin[L_];                                         // 12,288 B

  const int si   = blockIdx.x;   // 0..63 : Q super-row
  const int b    = blockIdx.y;   // 0..15 : batch
  const int tid  = threadIdx.x;  // 0..127
  const int w    = tid >> 5;     // wave id 0..3
  const int lane = tid & 31;
  const int m    = lane & 15;
  const int hi   = lane >> 4;    // 0 or 1
  const int aklo = hi << 3;      // A k-offset: 0 or 8  (ISA 7.12.2 layout)
  const int bklo = hi << 3;      // B k-offset: 0 or 8

  // zero the diagonal bins
  for (int t = tid; t < L_; t += 128) mbin[t] = 0.f;

  const unsigned short* __restrict__ Aq =
      Qbf + ((size_t)(b * L_ + 48 * si)) * C_;

#if defined(HAVE_TDM)
  if (tid < 32) {  // wave 0 issues the tensor DMA
    typedef unsigned int tv4u __attribute__((ext_vector_type(4)));
    typedef int tv8i __attribute__((ext_vector_type(8)));
    typedef int tv4i __attribute__((ext_vector_type(4)));
    unsigned lds_off = (unsigned)(uintptr_t)(void*)As;  // LDS aperture: low 32b
    unsigned long long ga = (unsigned long long)(uintptr_t)Aq;
    // D# group0: count=1 | lds_addr | global_addr[56:0] | type=2 (bits 127:126)
    tv4u g0 = { 1u, lds_off, (unsigned)ga,
                (unsigned)((ga >> 32) & 0x01ffffffu) | 0x80000000u };
    // D# group1: data_size=2B, pad_enable, pad_interval=256dw (code 7),
    //            pad_amount=4dw (code 3) -> LDS row stride 1040B = 520 elems.
    tv8i g1 = { (int)0x07D10000u,     // w0: mask=0|ds=1|pad_en|int=7|amt=3
                (int)(512u << 16),    // w1: tensor_dim0 = 512
                (int)(3072u << 16),   // w2: tensor_dim1 = 3072
                (int)(512u << 16),    // w3: tile_dim0 = 512
                48,                   // w4: tile_dim1 = 48 rows
                512, 0, 0 };          // w5: tensor_dim0_stride = 512
    tv4i g2 = {0, 0, 0, 0};           // 2-D tensor: groups 2/3 unused
    tv4i g3 = {0, 0, 0, 0};
#  if defined(TDM_6ARG)
    tv8i g4 = {0, 0, 0, 0, 0, 0, 0, 0};
    __builtin_amdgcn_tensor_load_to_lds(g0, g1, g2, g3, g4, 0);
#  else
    __builtin_amdgcn_tensor_load_to_lds(g0, g1, g2, g3, 0);
#  endif
    __builtin_amdgcn_s_wait_tensorcnt(0);
  }
#else
  // Fallback: cooperative copy into the same padded layout.
  for (int e = tid * 8; e < SI_TILES * 16 * C_; e += 128 * 8) {
    int row = e >> 9, col = e & (C_ - 1);
    *(uint4*)&As[row * APAD + col] = *(const uint4*)(Aq + (size_t)row * C_ + col);
  }
#endif
  __syncthreads();

  union V32 { uint4 u[2]; v16bf v; };
  const v8f vzero = {0.f, 0.f, 0.f, 0.f, 0.f, 0.f, 0.f, 0.f};

  for (int sjg = 0; sjg < 24; ++sjg) {
    const int Jt0 = 8 * sjg + 2 * w;  // this wave's first K tile
    const unsigned short* __restrict__ krow0 =
        Kbf + ((size_t)(b * L_ + (Jt0 << 4) + m)) * C_;
    const unsigned short* __restrict__ krow1 = krow0 + (size_t)16 * C_;
    __builtin_prefetch(krow0 + (size_t)128 * C_, 0, 1);  // next J-group

    v8f acc[4];  // g = (i - jj) + 1, i in 0..2 (Q tile), jj in 0..1 (K tile)
#pragma unroll
    for (int g = 0; g < 4; ++g) acc[g] = vzero;

#pragma unroll 2
    for (int kc = 0; kc < 16; ++kc) {
      const int c0 = kc << 5;  // 32 channels per WMMA step
      V32 b0, b1, a0, a1, a2;
      b0.u[0] = *(const uint4*)(krow0 + c0 + bklo);
      b0.u[1] = *(const uint4*)(krow0 + c0 + bklo + 16);
      b1.u[0] = *(const uint4*)(krow1 + c0 + bklo);
      b1.u[1] = *(const uint4*)(krow1 + c0 + bklo + 16);
      const unsigned short* ar = &As[m * APAD + c0 + aklo];
      a0.u[0] = *(const uint4*)(ar);
      a0.u[1] = *(const uint4*)(ar + 16);
      a1.u[0] = *(const uint4*)(ar + 16 * APAD);
      a1.u[1] = *(const uint4*)(ar + 16 * APAD + 16);
      a2.u[0] = *(const uint4*)(ar + 32 * APAD);
      a2.u[1] = *(const uint4*)(ar + 32 * APAD + 16);
      acc[1] = __builtin_amdgcn_wmma_f32_16x16x32_bf16(false, a0.v, false, b0.v,
                                                       (short)0, acc[1], false, false);
      acc[0] = __builtin_amdgcn_wmma_f32_16x16x32_bf16(false, a0.v, false, b1.v,
                                                       (short)0, acc[0], false, false);
      acc[2] = __builtin_amdgcn_wmma_f32_16x16x32_bf16(false, a1.v, false, b0.v,
                                                       (short)0, acc[2], false, false);
      acc[1] = __builtin_amdgcn_wmma_f32_16x16x32_bf16(false, a1.v, false, b1.v,
                                                       (short)0, acc[1], false, false);
      acc[3] = __builtin_amdgcn_wmma_f32_16x16x32_bf16(false, a2.v, false, b0.v,
                                                       (short)0, acc[3], false, false);
      acc[2] = __builtin_amdgcn_wmma_f32_16x16x32_bf16(false, a2.v, false, b1.v,
                                                       (short)0, acc[2], false, false);
    }

    // Scatter diagonal sums into LDS bins.  dd = Itile - Jtile = ddbase + g-1.
    const int ddbase = 3 * si - 8 * sjg - 2 * w;
#pragma unroll
    for (int g = 0; g < 4; ++g) {
      int dd = ddbase + g - 1;
      if (dd < 0) dd += NT;
#pragma unroll
      for (int r8 = 0; r8 < 8; ++r8) {
        int tau = (dd << 4) + (hi << 3) + r8 - m;  // lane holds column n = m
        if (tau < 0) tau += L_;
        atomicAdd(&mbin[tau], acc[g][r8]);
      }
    }
  }

  __syncthreads();
  // Flush block-local bins to global mean_value (corr.mean over H,E -> /512).
  const float scale = 1.0f / (float)C_;
  float* __restrict__ mrow = meanv + b * L_;
  for (int t = tid; t < L_; t += 128) atomicAdd(mrow + t, mbin[t] * scale);
}

// ---------------------------------------------------------------------------
// Kernel 3: batch-mean over b, then iterative top-8 argmax (single block).
// ---------------------------------------------------------------------------
__global__ void topk_kernel(const float* __restrict__ meanv,
                            int* __restrict__ index) {
  __shared__ float sm[L_];
  __shared__ float sval[256];
  __shared__ int   sidx[256];
  const int tid = threadIdx.x;

  for (int t = tid; t < L_; t += 256) {
    float s = 0.f;
#pragma unroll
    for (int b = 0; b < B_; ++b) s += meanv[b * L_ + t];
    sm[t] = s;  // scale irrelevant for argmax
  }
  __syncthreads();

  for (int it = 0; it < TOPK; ++it) {
    float best = -3.4e38f;
    int   bi   = 0;
    for (int t = tid; t < L_; t += 256) {
      float v = sm[t];
      if (v > best) { best = v; bi = t; }
    }
    sval[tid] = best;
    sidx[tid] = bi;
    __syncthreads();
    for (int ofs = 128; ofs > 0; ofs >>= 1) {
      if (tid < ofs) {
        if (sval[tid + ofs] > sval[tid] ||
            (sval[tid + ofs] == sval[tid] && sidx[tid + ofs] < sidx[tid])) {
          sval[tid] = sval[tid + ofs];
          sidx[tid] = sidx[tid + ofs];
        }
      }
      __syncthreads();
    }
    if (tid == 0) {
      index[it] = sidx[0];
      sm[sidx[0]] = -3.4e38f;  // mask out for next round
    }
    __syncthreads();
  }
}

// ---------------------------------------------------------------------------
// Kernel 4: per-batch softmax over the 8 selected lags.
// ---------------------------------------------------------------------------
__global__ void softmax_kernel(const float* __restrict__ meanv,
                               const int* __restrict__ index,
                               float* __restrict__ w) {
  int b = threadIdx.x;
  if (b >= B_) return;
  float v[TOPK];
  float m = -3.4e38f;
#pragma unroll
  for (int i = 0; i < TOPK; ++i) {
    v[i] = meanv[b * L_ + index[i]];
    m = fmaxf(m, v[i]);
  }
  float s = 0.f;
#pragma unroll
  for (int i = 0; i < TOPK; ++i) {
    v[i] = __expf(v[i] - m);
    s += v[i];
  }
  float inv = 1.0f / s;
#pragma unroll
  for (int i = 0; i < TOPK; ++i) w[b * TOPK + i] = v[i] * inv;
}

// ---------------------------------------------------------------------------
// Kernel 5: weighted circular gather.
// out[b,l,h,e] = sum_i w[b,i] * V[b, (l+index[i])%L, h, e]; float4 per thread.
// ---------------------------------------------------------------------------
__global__ void gather_kernel(const float* __restrict__ V,
                              const int* __restrict__ index,
                              const float* __restrict__ w,
                              float* __restrict__ out) {
  size_t flat = (size_t)blockIdx.x * blockDim.x + threadIdx.x;  // float4 units
  int cq = (int)(flat & 127);  // channel quad (C/4 = 128)
  int t  = (int)(flat >> 7);
  int l  = t % L_;
  int b  = t / L_;
  float4 acc = {0.f, 0.f, 0.f, 0.f};
#pragma unroll
  for (int i = 0; i < TOPK; ++i) {
    int src = l + index[i];
    if (src >= L_) src -= L_;
    const float4 vv = *(const float4*)(V + ((size_t)(b * L_ + src) * C_) + cq * 4);
    float wi = w[b * TOPK + i];
    acc.x += wi * vv.x;
    acc.y += wi * vv.y;
    acc.z += wi * vv.z;
    acc.w += wi * vv.w;
  }
  *(float4*)(out + flat * 4) = acc;
}

// ---------------------------------------------------------------------------
// Host launcher. Workspace layout (bytes):
//   [0, 50331648)            Q bf16
//   [50331648, 100663296)    K bf16
//   [100663296, 100859904)   mean_value (B*L f32)
//   [100859904, +256)        top-8 indices (int)
//   [+256, +768)             softmax weights (B*8 f32)
// ---------------------------------------------------------------------------
extern "C" void kernel_launch(void* const* d_in, const int* in_sizes, int n_in,
                              void* d_out, int out_size, void* d_ws, size_t ws_size,
                              hipStream_t stream) {
  const float* Q = (const float*)d_in[0];
  const float* K = (const float*)d_in[1];
  const float* V = (const float*)d_in[2];
  float* out = (float*)d_out;

  char* ws = (char*)d_ws;
  const size_t BF_BYTES = (size_t)B_ * L_ * C_ * 2;
  unsigned short* Qbf = (unsigned short*)(ws);
  unsigned short* Kbf = (unsigned short*)(ws + BF_BYTES);
  float* meanv        = (float*)(ws + 2 * BF_BYTES);
  int*   index        = (int*)(ws + 2 * BF_BYTES + (size_t)B_ * L_ * 4);
  float* wsoft        = (float*)(ws + 2 * BF_BYTES + (size_t)B_ * L_ * 4 + 256);

  const int nconv = (B_ * L_ * C_) / 4 / 256;  // 24576 blocks per tensor
  cvt_bf16_kernel<<<dim3(nconv, 2), 256, 0, stream>>>(Q, K, Qbf, Kbf);

  hipMemsetAsync(meanv, 0, (size_t)B_ * L_ * 4, stream);

  corr_wmma_kernel<<<dim3(NSI, B_), 128, 0, stream>>>(Qbf, Kbf, meanv);

  topk_kernel<<<1, 256, 0, stream>>>(meanv, index);
  softmax_kernel<<<1, 32, 0, stream>>>(meanv, index, wsoft);

  const int ngather = (B_ * L_ * C_) / 4 / 256;  // 24576 blocks
  gather_kernel<<<ngather, 256, 0, stream>>>(V, index, wsoft, out);
}
